// SoftCap_17033840296335
// MI455X (gfx1250) — compile-verified, matrix operand order checked by reading, not run on hardware
//
#include <hip/hip_runtime.h>
#include <cstddef>
#include <cstdint>

// ---------------- problem constants ----------------
constexpr int cB  = 64;      // batch
constexpr int cTV = 40;      // video timesteps
constexpr int cF  = 2048;    // feature dim
constexpr int cH  = 1024;    // hidden
constexpr int cV  = 32000;   // vocab
constexpr int cNS = 30;      // caption steps
constexpr int cCAP = cNS + 1;

// ---------------- WMMA types ----------------
typedef __bf16 v16bf __attribute__((ext_vector_type(16)));
typedef float  v8f   __attribute__((ext_vector_type(8)));
typedef unsigned v8u __attribute__((ext_vector_type(8)));

static __device__ __forceinline__ unsigned short f2bf(float f) {
  union { float f; unsigned u; } x{f};
  unsigned u = x.u;
  unsigned r = (u + 0x7FFFu + ((u >> 16) & 1u)) >> 16;   // RNE
  return (unsigned short)r;
}
static __device__ __forceinline__ unsigned pack2bf(float lo, float hi) {
  return (unsigned)f2bf(lo) | ((unsigned)f2bf(hi) << 16);
}
static __device__ __forceinline__ float sigmoidf(float x) {
  return 1.0f / (1.0f + __expf(-x));
}

// 8 consecutive dwords (two b128 loads)
static __device__ __forceinline__ v8u ld8(const unsigned* __restrict__ p) {
  uint4 q0 = *(const uint4*)(p);
  uint4 q1 = *(const uint4*)(p + 4);
  v8u r; r[0]=q0.x; r[1]=q0.y; r[2]=q0.z; r[3]=q0.w;
         r[4]=q1.x; r[5]=q1.y; r[6]=q1.z; r[7]=q1.w;
  return r;
}
// A fragment: dwords rowp[kb..kb+3] and rowp[kb+8..kb+11] (rowp includes half*4)
static __device__ __forceinline__ v8u ldA(const unsigned* __restrict__ rowp, int kb) {
  uint4 q0 = *(const uint4*)(rowp + kb);
  uint4 q1 = *(const uint4*)(rowp + kb + 8);
  v8u r; r[0]=q0.x; r[1]=q0.y; r[2]=q0.z; r[3]=q0.w;
         r[4]=q1.x; r[5]=q1.y; r[6]=q1.z; r[7]=q1.w;
  return r;
}
static __device__ __forceinline__ v8f wm(v8u a, v8u b, v8f c) {
  return __builtin_amdgcn_wmma_f32_16x16x32_bf16(
      false, __builtin_bit_cast(v16bf, a),
      false, __builtin_bit_cast(v16bf, b),
      (short)0, c, false, false);
}

// ---------------- generic f32 -> bf16 convert ----------------
__global__ void k_conv_bf16(unsigned short* __restrict__ dst,
                            const float* __restrict__ src, int n) {
  int i = blockIdx.x * blockDim.x + threadIdx.x;
  if (i < n) dst[i] = f2bf(src[i]);
}

// ---------------- pack a K x N f32 weight into WMMA B-fragment tiles -------
// dst layout: for N-tile t, K-chunk kk (32 K's), 256 dwords ordered
//   [lane][j]; dword(lane,j) = packed bf16 { B[k,n], B[k+1,n] },
//   k = kk*32 + (j<4?0:16) + (lane>>4)*8 + (j&3)*2 , n = t*16 + (lane&15)
// mode 0: src0 row-major [K,N]  (B[k,n] = src0[k*N+n])
// mode 1: B[k,n] = k<K0 ? src0[n*K0+k] : src1[n*(K-K0)+(k-K0)]   (W.T concat)
__global__ void k_pack_b(unsigned* __restrict__ dst,
                         const float* __restrict__ src0,
                         const float* __restrict__ src1,
                         int K0, int K, int N, int mode) {
  int kkIters = K >> 5;
  long long total = (long long)(N >> 4) * kkIters * 256;
  long long idx = (long long)blockIdx.x * blockDim.x + threadIdx.x;
  if (idx >= total) return;
  int t    = (int)(idx / ((long long)kkIters * 256));
  int rem  = (int)(idx % ((long long)kkIters * 256));
  int kk   = rem >> 8;
  int q    = rem & 255;
  int lane = q >> 3;
  int j    = q & 7;
  int n    = t * 16 + (lane & 15);
  int k    = kk * 32 + ((j < 4) ? 0 : 16) + (lane >> 4) * 8 + (j & 3) * 2;
  float v0, v1;
  if (mode == 0) {
    v0 = src0[(size_t)k * N + n];
    v1 = src0[(size_t)(k + 1) * N + n];
  } else {                       // k even, K0 even => k and k+1 in same region
    int K1 = K - K0;
    if (k < K0) { v0 = src0[(size_t)n * K0 + k];        v1 = src0[(size_t)n * K0 + k + 1]; }
    else        { v0 = src1[(size_t)n * K1 + (k - K0)]; v1 = src1[(size_t)n * K1 + (k - K0) + 1]; }
  }
  dst[idx] = pack2bf(v0, v1);
}

// ---------------- bf16 WMMA GEMM: out[M,N] = A[M,K] * Bpk + bias ------------
// A: bf16 row-major as dwords (K/2 per row). Wave => 64(M) x 32(N) tile,
// double-buffered K pipeline. Requires N % 128 == 0, M % 64 == 0, K % 64 == 0.
// remapT>0: output row m scattered to (m%remapT)*remapB + (m/remapT).
__global__ __launch_bounds__(128)
void k_gemm_bf16(const unsigned* __restrict__ A,
                 const unsigned* __restrict__ Bpk,
                 const float* __restrict__ bias,
                 float* __restrict__ out,
                 int M, int K, int N, int remapT, int remapB) {
  const int lane = threadIdx.x & 31;
  const int wave = threadIdx.x >> 5;
  const int ntp  = blockIdx.x * 8 + wave * 2;          // two N tiles per wave
  if (ntp * 16 >= N) return;
  const int m0   = blockIdx.y * 64;
  const int Kd   = K >> 1;
  const int kkIt = K >> 5;
  const int half = lane >> 4;
  const int mloc = lane & 15;

  const unsigned* rowp[4];
  #pragma unroll
  for (int i = 0; i < 4; ++i)
    rowp[i] = A + (size_t)(m0 + i * 16 + mloc) * Kd + half * 4;

  const unsigned* b0 = Bpk + (size_t)ntp * kkIt * 256 + lane * 8;
  const unsigned* b1 = b0 + (size_t)kkIt * 256;

  v8f acc[2][4];
  #pragma unroll
  for (int n = 0; n < 2; ++n)
    #pragma unroll
    for (int i = 0; i < 4; ++i) acc[n][i] = (v8f){};

  v8u Ab[2][4], Bb[2][2];

  // prologue: fragments for kk = 0
  #pragma unroll
  for (int i = 0; i < 4; ++i) Ab[0][i] = ldA(rowp[i], 0);
  Bb[0][0] = ld8(b0); Bb[0][1] = ld8(b1);
  int kb = 16; b0 += 256; b1 += 256;

  // steady state: kkIt is even (K % 64 == 0)
  for (int kk = 0; kk < kkIt - 2; kk += 2) {
    // prefetch kk+1 into buffer 1
    #pragma unroll
    for (int i = 0; i < 4; ++i) Ab[1][i] = ldA(rowp[i], kb);
    Bb[1][0] = ld8(b0); Bb[1][1] = ld8(b1);
    kb += 16; b0 += 256; b1 += 256;
    // compute kk (buffer 0)
    #pragma unroll
    for (int i = 0; i < 4; ++i) {
      acc[0][i] = wm(Ab[0][i], Bb[0][0], acc[0][i]);
      acc[1][i] = wm(Ab[0][i], Bb[0][1], acc[1][i]);
    }
    // prefetch kk+2 into buffer 0
    #pragma unroll
    for (int i = 0; i < 4; ++i) Ab[0][i] = ldA(rowp[i], kb);
    Bb[0][0] = ld8(b0); Bb[0][1] = ld8(b1);
    kb += 16; b0 += 256; b1 += 256;
    // compute kk+1 (buffer 1)
    #pragma unroll
    for (int i = 0; i < 4; ++i) {
      acc[0][i] = wm(Ab[1][i], Bb[1][0], acc[0][i]);
      acc[1][i] = wm(Ab[1][i], Bb[1][1], acc[1][i]);
    }
  }
  // epilogue: last two K-chunks
  #pragma unroll
  for (int i = 0; i < 4; ++i) Ab[1][i] = ldA(rowp[i], kb);
  Bb[1][0] = ld8(b0); Bb[1][1] = ld8(b1);
  #pragma unroll
  for (int i = 0; i < 4; ++i) {
    acc[0][i] = wm(Ab[0][i], Bb[0][0], acc[0][i]);
    acc[1][i] = wm(Ab[0][i], Bb[0][1], acc[1][i]);
  }
  #pragma unroll
  for (int i = 0; i < 4; ++i) {
    acc[0][i] = wm(Ab[1][i], Bb[1][0], acc[0][i]);
    acc[1][i] = wm(Ab[1][i], Bb[1][1], acc[1][i]);
  }

  // ---- store D: lane holds (m = m0+i*16+j+half*8, n = nt*16 + mloc) ----
  #pragma unroll
  for (int nsub = 0; nsub < 2; ++nsub) {
    const int n = (ntp + nsub) * 16 + mloc;
    const float bv = bias ? bias[n] : 0.0f;
    #pragma unroll
    for (int i = 0; i < 4; ++i) {
      #pragma unroll
      for (int j = 0; j < 8; ++j) {
        int m = m0 + i * 16 + j + half * 8;
        int orow = remapT ? ((m % remapT) * remapB + (m / remapT)) : m;
        out[(size_t)orow * N + n] = acc[nsub][i][j] + bv;
      }
    }
  }
}

// ---------------- feat_mean over TV, plus bf16 mirror ----------------
__global__ void k_feat_mean(const float* __restrict__ act, float* __restrict__ fm,
                            unsigned short* __restrict__ fmbf) {
  int i = blockIdx.x * blockDim.x + threadIdx.x;     // over B*F
  if (i >= cB * cF) return;
  int b = i / cF, f = i % cF;
  float s = 0.f;
  for (int t = 0; t < cTV; ++t) s += act[((size_t)b * cTV + t) * cF + f];
  s *= (1.0f / cTV);
  fm[i] = s; fmbf[i] = f2bf(s);
}

// ---------------- initialize carries and loss ----------------
__global__ void k_init_state(const float* __restrict__ fmH,
                             float* h1, float* c1, float* h2, float* c2,
                             unsigned short* h2bf, float* loss) {
  int i = blockIdx.x * blockDim.x + threadIdx.x;
  if (i >= cB * cH) return;
  float v = fmH[i];
  h1[i] = v; c1[i] = 0.f; h2[i] = 0.f; c2[i] = 0.f;
  h2bf[i] = f2bf(v);                 // h_prev(0) = feat_mean
  if (i == 0) loss[0] = 0.f;
}

__global__ void k_bias_combine(float* dst, const float* a, const float* b, int n) {
  int i = blockIdx.x * blockDim.x + threadIdx.x;
  if (i < n) dst[i] = a[i] + b[i];
}

// ---------------- attention: e[t*B+b] = tanh(hWa[b]+ip[t,b]) . att_w -------
__global__ __launch_bounds__(256)
void k_attn_e(const float* __restrict__ hWa, const float* __restrict__ ip,
              const float* __restrict__ aw, float* __restrict__ e) {
  __shared__ float red[256];
  int tb = blockIdx.x;            // t*B + b
  int b  = tb % cB;
  float s = 0.f;
  for (int h = threadIdx.x; h < cH; h += 256)
    s += tanhf(hWa[(size_t)b * cH + h] + ip[(size_t)tb * cH + h]) * aw[h];
  red[threadIdx.x] = s; __syncthreads();
  for (int o = 128; o > 0; o >>= 1) {
    if (threadIdx.x < o) red[threadIdx.x] += red[threadIdx.x + o];
    __syncthreads();
  }
  if (threadIdx.x == 0) e[tb] = red[0];
}

// ---------------- masked softmax over T + weighted sum of vid_emb ----------
__global__ __launch_bounds__(256)
void k_attn_combine(const float* __restrict__ e, const float* __restrict__ vmask,
                    const float* __restrict__ vid, float* __restrict__ atten) {
  __shared__ float ediv[cTV];
  int b = blockIdx.x;
  if (threadIdx.x == 0) {
    float emax = -1e30f;
    for (int t = 0; t < cTV; ++t) emax = fmaxf(emax, e[t * cB + b]);
    float denom = 0.f;
    for (int t = 0; t < cTV; ++t) {
      float eh = vmask[(size_t)b * cTV + t] * __expf(e[t * cB + b] - emax);
      ediv[t] = eh; denom += eh;
    }
    denom = denom + ((denom == 0.0f) ? 1.0f : 0.0f) + 1e-9f;
    float inv = 1.0f / denom;
    for (int t = 0; t < cTV; ++t) ediv[t] *= inv;
  }
  __syncthreads();
  for (int h = threadIdx.x; h < cH; h += 256) {
    float s = 0.f;
    for (int t = 0; t < cTV; ++t)
      s += ediv[t] * vid[((size_t)t * cB + b) * cH + h];
    atten[(size_t)b * cH + h] = s;
  }
}

// ---------------- build x1 = [emb_t | atten | h1] as bf16 [B, 3H] ----------
__global__ void k_build_x1(const float* __restrict__ Wemb, const int* __restrict__ cap,
                           int step, const float* __restrict__ atten,
                           const float* __restrict__ h1,
                           unsigned short* __restrict__ x1bf) {
  int i = blockIdx.x * blockDim.x + threadIdx.x;     // over B * 3H
  if (i >= cB * 3 * cH) return;
  int b = i / (3 * cH), c = i % (3 * cH);
  float v;
  if (c < cH) {
    int tok = cap[(size_t)b * cCAP + step];
    v = Wemb[(size_t)tok * cH + c];
  } else if (c < 2 * cH) {
    v = atten[(size_t)b * cH + (c - cH)];
  } else {
    v = h1[(size_t)b * cH + (c - 2 * cH)];
  }
  x1bf[i] = f2bf(v);
}

// ---------------- LSTM cell 1: update h1/c1; build A2 = [h1 | h2] bf16 -----
__global__ void k_lstm_cell1(const float* __restrict__ g, float* c1, float* h1,
                             const float* __restrict__ h2,
                             unsigned short* __restrict__ a2bf) {
  int i = blockIdx.x * blockDim.x + threadIdx.x;
  if (i >= cB * cH) return;
  int b = i / cH, h = i % cH;
  const float* gr = g + (size_t)b * 4 * cH;
  float ig = gr[h], fg = gr[cH + h], gg = gr[2 * cH + h], og = gr[3 * cH + h];
  float cn = sigmoidf(fg) * c1[i] + sigmoidf(ig) * tanhf(gg);
  float hn = sigmoidf(og) * tanhf(cn);
  c1[i] = cn; h1[i] = hn;
  a2bf[(size_t)b * 2 * cH + h]      = f2bf(hn);
  a2bf[(size_t)b * 2 * cH + cH + h] = f2bf(h2[i]);
}

// ---------------- LSTM cell 2: update h2/c2; bf16 mirror (h_prev & logits A)
__global__ void k_lstm_cell2(const float* __restrict__ g, float* c2, float* h2,
                             unsigned short* __restrict__ h2bf) {
  int i = blockIdx.x * blockDim.x + threadIdx.x;
  if (i >= cB * cH) return;
  int b = i / cH, h = i % cH;
  const float* gr = g + (size_t)b * 4 * cH;
  float ig = gr[h], fg = gr[cH + h], gg = gr[2 * cH + h], og = gr[3 * cH + h];
  float cn = sigmoidf(fg) * c2[i] + sigmoidf(ig) * tanhf(gg);
  float hn = sigmoidf(og) * tanhf(cn);
  c2[i] = cn; h2[i] = hn; h2bf[i] = f2bf(hn);
}

// ---------------- per-row cross entropy over V ----------------
__global__ __launch_bounds__(256)
void k_ce_row(const float* __restrict__ logits, const int* __restrict__ cap,
              int step, float* __restrict__ rowloss) {
  __shared__ float red[256];
  __shared__ float rmax;
  int b = blockIdx.x;
  const float* lr = logits + (size_t)b * cV;
  float m = -1e30f;
  for (int j = threadIdx.x; j < cV; j += 256) m = fmaxf(m, lr[j]);
  red[threadIdx.x] = m; __syncthreads();
  for (int o = 128; o > 0; o >>= 1) {
    if (threadIdx.x < o) red[threadIdx.x] = fmaxf(red[threadIdx.x], red[threadIdx.x + o]);
    __syncthreads();
  }
  if (threadIdx.x == 0) rmax = red[0];
  __syncthreads();
  float s = 0.f;
  for (int j = threadIdx.x; j < cV; j += 256) s += __expf(lr[j] - rmax);
  red[threadIdx.x] = s; __syncthreads();
  for (int o = 128; o > 0; o >>= 1) {
    if (threadIdx.x < o) red[threadIdx.x] += red[threadIdx.x + o];
    __syncthreads();
  }
  if (threadIdx.x == 0) {
    int tgt = cap[(size_t)b * cCAP + step + 1];
    rowloss[b] = -(lr[tgt] - rmax - __logf(red[0]));
  }
}

__global__ void k_ce_acc(const float* __restrict__ rowloss,
                         const float* __restrict__ cmask, int step,
                         float* __restrict__ loss) {
  __shared__ float rl[64], mk[64];
  int t = threadIdx.x;
  rl[t] = rowloss[t];
  mk[t] = cmask[(size_t)t * cCAP + step + 1];
  __syncthreads();
  if (t == 0) {
    float sl = 0.f, sm = 0.f;
    for (int i = 0; i < cB; ++i) { sl += rl[i]; sm += mk[i]; }
    loss[0] += (sl / (float)cB) * sm / (float)cB;
  }
}

__global__ void k_write_out(const float* loss, float* out) {
  if (threadIdx.x == 0 && blockIdx.x == 0) out[0] = loss[0];
}

// ====================== host side ======================
static inline size_t alignup(size_t x) { return (x + 255) & ~size_t(255); }

extern "C" void kernel_launch(void* const* d_in, const int* in_sizes, int n_in,
                              void* d_out, int out_size, void* d_ws, size_t ws_size,
                              hipStream_t stream) {
  (void)in_sizes; (void)n_in; (void)out_size; (void)ws_size;
  const float* action_feat = (const float*)d_in[1];
  const float* video_mask  = (const float*)d_in[3];
  const int*   captions    = (const int*)d_in[4];
  const float* cap_mask    = (const float*)d_in[5];
  const float* enc_img_W   = (const float*)d_in[6];
  const float* enc_img_b   = (const float*)d_in[7];
  const float* enc_mean_W  = (const float*)d_in[8];
  const float* enc_mean_b  = (const float*)d_in[9];
  const float* att_w       = (const float*)d_in[10];
  const float* att_Wa      = (const float*)d_in[11];
  const float* att_Ua      = (const float*)d_in[12];
  const float* att_ba      = (const float*)d_in[13];
  const float* Wemb        = (const float*)d_in[14];
  const float* embed_W     = (const float*)d_in[15];
  const float* embed_b     = (const float*)d_in[16];
  const float* l1_Wih = (const float*)d_in[17];
  const float* l1_Whh = (const float*)d_in[18];
  const float* l1_bih = (const float*)d_in[19];
  const float* l1_bhh = (const float*)d_in[20];
  const float* l2_Wih = (const float*)d_in[21];
  const float* l2_Whh = (const float*)d_in[22];
  const float* l2_bih = (const float*)d_in[23];
  const float* l2_bhh = (const float*)d_in[24];

  // ---- bump-allocate workspace ----
  char* p = (char*)d_ws;
  auto alloc = [&](size_t bytes) { char* r = p; p += alignup(bytes); return r; };
  unsigned short* act_bf   = (unsigned short*)alloc((size_t)cB * cTV * cF * 2);
  unsigned* enc_img_pk     = (unsigned*)alloc((size_t)(cH/16)*(cF/32)*256*4);
  unsigned* enc_mean_pk    = (unsigned*)alloc((size_t)(cH/16)*(cF/32)*256*4);
  unsigned* attWa_pk       = (unsigned*)alloc((size_t)(cH/16)*(cH/32)*256*4);
  unsigned* attUa_pk       = (unsigned*)alloc((size_t)(cH/16)*(cH/32)*256*4);
  unsigned* B1pk           = (unsigned*)alloc((size_t)(4*cH/16)*(3*cH/32)*256*4);
  unsigned* B2pk           = (unsigned*)alloc((size_t)(4*cH/16)*(2*cH/32)*256*4);
  unsigned* Wlog_pk        = (unsigned*)alloc((size_t)(cV/16)*(cH/32)*256*4);
  float* vid_emb           = (float*)alloc((size_t)cTV * cB * cH * 4);
  unsigned short* vid_bf   = (unsigned short*)alloc((size_t)cTV * cB * cH * 2);
  float* image_part        = (float*)alloc((size_t)cTV * cB * cH * 4);
  float* fm                = (float*)alloc((size_t)cB * cF * 4);
  unsigned short* fm_bf    = (unsigned short*)alloc((size_t)cB * cF * 2);
  float* fmH               = (float*)alloc((size_t)cB * cH * 4);
  float* bias1             = (float*)alloc(4 * cH * 4);
  float* bias2             = (float*)alloc(4 * cH * 4);
  float* h1 = (float*)alloc((size_t)cB*cH*4);
  float* c1 = (float*)alloc((size_t)cB*cH*4);
  float* h2 = (float*)alloc((size_t)cB*cH*4);
  float* c2 = (float*)alloc((size_t)cB*cH*4);
  unsigned short* h2bf = (unsigned short*)alloc((size_t)cB*cH*2);
  float* hWa   = (float*)alloc((size_t)cB*cH*4);
  float* evec  = (float*)alloc((size_t)cTV*cB*4);
  float* atten = (float*)alloc((size_t)cB*cH*4);
  unsigned short* x1bf = (unsigned short*)alloc((size_t)cB*3*cH*2);
  unsigned short* a2bf = (unsigned short*)alloc((size_t)cB*2*cH*2);
  float* gates1 = (float*)alloc((size_t)cB*4*cH*4);
  float* gates2 = (float*)alloc((size_t)cB*4*cH*4);
  float* logits = (float*)alloc((size_t)cB*cV*4);
  float* rowloss = (float*)alloc(cB*4);
  float* lossacc = (float*)alloc(256);

  auto blocks = [](long long n, int bs) { return (unsigned)((n + bs - 1) / bs); };
  auto packN  = [](int K, int N) { return (long long)(N/16) * (K/32) * 256; };

  // ---------- precompute ----------
  k_conv_bf16<<<blocks((long long)cB*cTV*cF,256),256,0,stream>>>(act_bf, action_feat, cB*cTV*cF);
  k_pack_b<<<blocks(packN(cF,cH),256),256,0,stream>>>(enc_img_pk,  enc_img_W,  nullptr, 0, cF, cH, 0);
  k_pack_b<<<blocks(packN(cF,cH),256),256,0,stream>>>(enc_mean_pk, enc_mean_W, nullptr, 0, cF, cH, 0);
  k_pack_b<<<blocks(packN(cH,cH),256),256,0,stream>>>(attWa_pk, att_Wa, nullptr, 0, cH, cH, 0);
  k_pack_b<<<blocks(packN(cH,cH),256),256,0,stream>>>(attUa_pk, att_Ua, nullptr, 0, cH, cH, 0);
  k_pack_b<<<blocks(packN(3*cH,4*cH),256),256,0,stream>>>(B1pk, l1_Wih, l1_Whh, 2*cH, 3*cH, 4*cH, 1);
  k_pack_b<<<blocks(packN(2*cH,4*cH),256),256,0,stream>>>(B2pk, l2_Wih, l2_Whh, cH,   2*cH, 4*cH, 1);
  k_pack_b<<<blocks(packN(cH,cV),256),256,0,stream>>>(Wlog_pk, embed_W, nullptr, 0, cH, cV, 0);
  k_bias_combine<<<blocks(4*cH,256),256,0,stream>>>(bias1, l1_bih, l1_bhh, 4*cH);
  k_bias_combine<<<blocks(4*cH,256),256,0,stream>>>(bias2, l2_bih, l2_bhh, 4*cH);

  // vid_emb[t,b,:] = (action_feat @ enc_img_W + b), rows remapped b*TV+t -> t*B+b
  k_gemm_bf16<<<dim3(cH/128, (cB*cTV)/64), 128, 0, stream>>>(
      (const unsigned*)act_bf, enc_img_pk, enc_img_b, vid_emb,
      cB*cTV, cF, cH, cTV, cB);
  k_feat_mean<<<blocks((long long)cB*cF,256),256,0,stream>>>(action_feat, fm, fm_bf);
  k_gemm_bf16<<<dim3(cH/128, 1), 128, 0, stream>>>(
      (const unsigned*)fm_bf, enc_mean_pk, enc_mean_b, fmH, cB, cF, cH, 0, 0);
  k_init_state<<<blocks((long long)cB*cH,256),256,0,stream>>>(fmH, h1, c1, h2, c2, h2bf, lossacc);
  k_conv_bf16<<<blocks((long long)cTV*cB*cH,256),256,0,stream>>>(vid_bf, vid_emb, cTV*cB*cH);
  // image_part = vid_emb @ att_Ua + att_ba   (rows already t*B+b)
  k_gemm_bf16<<<dim3(cH/128, (cTV*cB)/64), 128, 0, stream>>>(
      (const unsigned*)vid_bf, attUa_pk, att_ba, image_part, cTV*cB, cH, cH, 0, 0);

  // ---------- recurrent steps ----------
  for (int s = 0; s < cNS; ++s) {
    // hWa = h_prev @ att_Wa   (h_prev bf16 lives in h2bf)
    k_gemm_bf16<<<dim3(cH/128, 1), 128, 0, stream>>>(
        (const unsigned*)h2bf, attWa_pk, nullptr, hWa, cB, cH, cH, 0, 0);
    k_attn_e<<<cTV*cB, 256, 0, stream>>>(hWa, image_part, att_w, evec);
    k_attn_combine<<<cB, 256, 0, stream>>>(evec, video_mask, vid_emb, atten);
    k_build_x1<<<blocks((long long)cB*3*cH,256),256,0,stream>>>(Wemb, captions, s, atten, h1, x1bf);
    // gates1 = [emb|atten|h1] @ [Wih1^T ; Whh1^T] + (bih1+bhh1)
    k_gemm_bf16<<<dim3((4*cH)/128, 1), 128, 0, stream>>>(
        (const unsigned*)x1bf, B1pk, bias1, gates1, cB, 3*cH, 4*cH, 0, 0);
    k_lstm_cell1<<<blocks((long long)cB*cH,256),256,0,stream>>>(gates1, c1, h1, h2, a2bf);
    // gates2 = [h1|h2] @ [Wih2^T ; Whh2^T] + (bih2+bhh2)
    k_gemm_bf16<<<dim3((4*cH)/128, 1), 128, 0, stream>>>(
        (const unsigned*)a2bf, B2pk, bias2, gates2, cB, 2*cH, 4*cH, 0, 0);
    k_lstm_cell2<<<blocks((long long)cB*cH,256),256,0,stream>>>(gates2, c2, h2, h2bf);
    // logits = h2 @ embed_word_W + embed_word_b
    k_gemm_bf16<<<dim3(cV/128, 1), 128, 0, stream>>>(
        (const unsigned*)h2bf, Wlog_pk, embed_b, logits, cB, cH, cV, 0, 0);
    k_ce_row<<<cB, 256, 0, stream>>>(logits, captions, s, rowloss);
    k_ce_acc<<<1, 64, 0, stream>>>(rowloss, cap_mask, s, lossacc);
  }
  k_write_out<<<1, 32, 0, stream>>>(lossacc, (float*)d_out);
}
